// ModularMeta_18562848654163
// MI455X (gfx1250) — compile-verified, matrix operand order checked by reading, not run on hardware
//
#include <hip/hip_runtime.h>
#include <hip/hip_bf16.h>

// ---------------------------------------------------------------------------
// Types for CDNA5 WMMA (gfx1250, wave32)
// ---------------------------------------------------------------------------
typedef __attribute__((ext_vector_type(16))) __bf16   v16bf;
typedef __attribute__((ext_vector_type(8)))  float    v8f;
typedef __attribute__((ext_vector_type(4)))  unsigned int v4u;

union AFrag {
    v16bf v;
    v4u   q[2];
    unsigned short s[16];
};

__device__ __forceinline__ unsigned short f32_to_bf16_bits(float x) {
    unsigned int u = __float_as_uint(x);
    return (unsigned short)((u + 0x7FFFu + ((u >> 16) & 1u)) >> 16);
}

// ---------------------------------------------------------------------------
// Attention head: direct 3x3 conv (SAME, stride 1), fp32
// ---------------------------------------------------------------------------
__global__ void conv3x3_f32(const float* __restrict__ x, const float* __restrict__ w,
                            const float* __restrict__ bias, float* __restrict__ y,
                            int B, int Cin, int Cout, int H, int W)
{
    long idx = (long)blockIdx.x * blockDim.x + threadIdx.x;
    long total = (long)B * Cout * H * W;
    if (idx >= total) return;
    int wx = (int)(idx % W); long t = idx / W;
    int hy = (int)(t % H);   t /= H;
    int o  = (int)(t % Cout);
    int b  = (int)(t / Cout);

    float acc = bias[o];
    const float* wo = w + (size_t)o * Cin * 9;
    const float* xb = x + (size_t)b * Cin * H * W;
    for (int i = 0; i < Cin; ++i) {
        const float* xc = xb + (size_t)i * H * W;
        const float* wc = wo + i * 9;
        #pragma unroll
        for (int r = 0; r < 3; ++r) {
            int yy = hy + r - 1;
            if (yy < 0 || yy >= H) continue;
            #pragma unroll
            for (int c = 0; c < 3; ++c) {
                int xx = wx + c - 1;
                if (xx < 0 || xx >= W) continue;
                acc += wc[r * 3 + c] * xc[yy * W + xx];
            }
        }
    }
    y[idx] = acc;
}

// ---------------------------------------------------------------------------
// BatchNorm batch statistics: one block per channel
// ---------------------------------------------------------------------------
__global__ void bn_stats(const float* __restrict__ x, float* __restrict__ stats,
                         int B, int C, int HW)
{
    int c = blockIdx.x;
    int tid = threadIdx.x;
    float s = 0.f, s2 = 0.f;
    long n = (long)B * HW;
    for (long e = tid; e < n; e += 256) {
        long b = e / HW, p = e % HW;
        float v = x[((size_t)b * C + c) * HW + p];
        s += v; s2 += v * v;
    }
    __shared__ float rs[256], rs2[256];
    rs[tid] = s; rs2[tid] = s2;
    __syncthreads();
    for (int st = 128; st > 0; st >>= 1) {
        if (tid < st) { rs[tid] += rs[tid + st]; rs2[tid] += rs2[tid + st]; }
        __syncthreads();
    }
    if (tid == 0) {
        float m   = rs[0] / (float)n;
        float var = rs2[0] / (float)n - m * m;
        stats[c * 2]     = m;
        stats[c * 2 + 1] = rsqrtf(var + 1e-5f);
    }
}

// ---------------------------------------------------------------------------
// Fused normalize + LeakyReLU + 2x2 maxpool
// ---------------------------------------------------------------------------
__global__ void bn_lrelu_pool(const float* __restrict__ x, const float* __restrict__ stats,
                              const float* __restrict__ g, const float* __restrict__ bt,
                              float* __restrict__ y, int B, int C, int H, int W)
{
    int H2 = H >> 1, W2 = W >> 1;
    long idx = (long)blockIdx.x * blockDim.x + threadIdx.x;
    long total = (long)B * C * H2 * W2;
    if (idx >= total) return;
    int wo = (int)(idx % W2); long t = idx / W2;
    int ho = (int)(t % H2);   t /= H2;
    int c  = (int)(t % C);
    int b  = (int)(t / C);

    float m = stats[c * 2], inv = stats[c * 2 + 1], gg = g[c], bb = bt[c];
    const float* xp = x + (((size_t)b * C + c) * H + ho * 2) * W + wo * 2;
    float best = -3.4e38f;
    #pragma unroll
    for (int r = 0; r < 2; ++r)
        #pragma unroll
        for (int cc = 0; cc < 2; ++cc) {
            float v = gg * (xp[r * W + cc] - m) * inv + bb;
            v = (v >= 0.f) ? v : 0.01f * v;
            best = fmaxf(best, v);
        }
    y[idx] = best;
}

// ---------------------------------------------------------------------------
// Linear [4096 -> 8] + softmax, one block per sample
// ---------------------------------------------------------------------------
__global__ void linear_softmax(const float* __restrict__ a, const float* __restrict__ lw,
                               const float* __restrict__ lb, float* __restrict__ f)
{
    int b = blockIdx.x, tid = threadIdx.x;
    __shared__ float red[8 * 256];
    const float* ab = a + (size_t)b * 4096;
    #pragma unroll
    for (int t = 0; t < 8; ++t) {
        float s = 0.f;
        const float* wr = lw + (size_t)t * 4096;
        for (int j = tid; j < 4096; j += 256) s += ab[j] * wr[j];
        red[t * 256 + tid] = s;
    }
    __syncthreads();
    for (int st = 128; st > 0; st >>= 1) {
        if (tid < st) {
            #pragma unroll
            for (int t = 0; t < 8; ++t) red[t * 256 + tid] += red[t * 256 + tid + st];
        }
        __syncthreads();
    }
    if (tid == 0) {
        float lg[8], mx = -3.4e38f;
        #pragma unroll
        for (int t = 0; t < 8; ++t) { lg[t] = red[t * 256] + lb[t]; mx = fmaxf(mx, lg[t]); }
        float sum = 0.f;
        #pragma unroll
        for (int t = 0; t < 8; ++t) { lg[t] = __expf(lg[t] - mx); sum += lg[t]; }
        #pragma unroll
        for (int t = 0; t < 8; ++t) f[b * 8 + t] = lg[t] / sum;
    }
}

// ---------------------------------------------------------------------------
// Mix expert weights: mw[b][o][k] = sum_t f[b][t] * W[t][o][k] (bf16, padded)
// ---------------------------------------------------------------------------
__global__ void mix_weights(const float* __restrict__ f, const float* __restrict__ W,
                            const float* __restrict__ Bv,
                            unsigned short* __restrict__ mw, float* __restrict__ mb,
                            int Bn, int Cout, int Kreal, int Mp, int KP)
{
    long idx = (long)blockIdx.x * blockDim.x + threadIdx.x;
    long total = (long)Bn * Mp * KP;
    if (idx >= total) return;
    int k = (int)(idx % KP); long t = idx / KP;
    int o = (int)(t % Mp);
    int b = (int)(t / Mp);
    float s = 0.f;
    const float* fb = f + b * 8;
    if (o < Cout && k < Kreal) {
        #pragma unroll
        for (int tt = 0; tt < 8; ++tt)
            s += fb[tt] * W[((size_t)tt * Cout + o) * Kreal + k];
    }
    mw[idx] = f32_to_bf16_bits(s);
    if (k == 0) {
        float sb = 0.f;
        if (o < Cout) {
            #pragma unroll
            for (int tt = 0; tt < 8; ++tt) sb += fb[tt] * Bv[tt * Cout + o];
        }
        mb[(size_t)b * Mp + o] = sb;
    }
}

// ---------------------------------------------------------------------------
// fp32 -> bf16 conversion
// ---------------------------------------------------------------------------
__global__ void cast_bf16(const float* __restrict__ x, unsigned short* __restrict__ y, long n)
{
    long i = (long)blockIdx.x * blockDim.x + threadIdx.x;
    if (i < n) y[i] = f32_to_bf16_bits(x[i]);
}

// ---------------------------------------------------------------------------
// Per-sample dynamic 3x3 conv as implicit GEMM on v_wmma_f32_16x16x32_bf16.
//   M = out channels (MCHUNKS x 16), K = CIN*9 padded to KP, N = pixels.
// Block = 1 sample x 2 image rows (128 px). 8 waves, each owns a 16-px N-tile
// and iterates all M-chunks so the LDS B-gather is amortized over MCHUNKS WMMAs.
//
// Halo patch: 4 rows x CIN channels staged in LDS with padded row width 80
// (image col x lives at index x+8; guard columns 7 and 72 are zero), so the
// hot-loop gather index  i*320 + (ly+r)*80 + (x+c+7)  needs no bounds checks.
// Interior rows are staged with CDNA5 async global->LDS B128 transfers
// (ASYNCcnt), guard columns / out-of-image rows are zero-filled with DS stores.
// ---------------------------------------------------------------------------
template <int CIN, int KP, int MCHUNKS, bool LRELU, bool OUTF32>
__global__ __launch_bounds__(256, 2)
void dyn_conv_wmma(const unsigned short* __restrict__ act,   // [B][CIN][64][64] bf16 bits
                   const unsigned short* __restrict__ mw,    // [B][MCHUNKS*16][KP] bf16 bits
                   const float* __restrict__ mb,             // [B][MCHUNKS*16]
                   unsigned short* __restrict__ out_bf,      // [B][64][64][64] bf16 bits
                   float* __restrict__ out_f)                // [B][2][64][64] fp32
{
    constexpr int MP   = MCHUNKS * 16;
    constexpr int ROWW = 80;                 // padded row width (160B, 16B aligned chunks)
    const int b    = blockIdx.x;
    const int y0   = blockIdx.y * 2;
    const int tid  = threadIdx.x;
    const int wave = tid >> 5;
    const int lane = tid & 31;
    const int lo16 = lane & 15;
    const int hi   = lane >> 4;

    __shared__ alignas(16) unsigned short patch[CIN * 4 * ROWW];

    const size_t actBase = (size_t)b * CIN * 4096;

    // Zero the guard columns of every (channel, row) line.
    for (int e = tid; e < CIN * 4; e += 256) {
        patch[e * ROWW + 7]  = 0;
        patch[e * ROWW + 72] = 0;
    }

    // Stage interior pixels: 8 chunks of 8 bf16 (16B) per line.
    // In-range lines use async global->LDS B128; out-of-image lines are zeroed.
    for (int e = tid; e < CIN * 4 * 8; e += 256) {
        int chunk = e & 7;
        int line  = e >> 3;          // i*4 + pr
        int pr    = line & 3;
        int i     = line >> 2;
        int gy    = y0 - 1 + pr;
        unsigned short* ldst = &patch[line * ROWW + 8 + chunk * 8];
        if (gy >= 0 && gy < 64) {
            const unsigned short* gsrc = act + actBase + (size_t)i * 4096 + gy * 64 + chunk * 8;
            unsigned int       ldsoff = (unsigned int)(unsigned long long)(uintptr_t)ldst;
            unsigned long long gaddr  = (unsigned long long)(uintptr_t)gsrc;
            asm volatile("global_load_async_to_lds_b128 %0, %1, off"
                         :: "v"(ldsoff), "v"(gaddr) : "memory");
        } else {
            #pragma unroll
            for (int q = 0; q < 8; ++q) ldst[q] = 0;
        }
    }
    asm volatile("s_wait_asynccnt 0x0" ::: "memory");
    __syncthreads();

    // wave -> 16 output pixels within the 2-row strip
    const int p0 = wave * 16;
    const int ly = p0 >> 6;         // local output row (0/1)
    const int yr = y0 + ly;
    const int x  = (p0 & 63) + lo16;

    v8f acc[MCHUNKS];
    #pragma unroll
    for (int m = 0; m < MCHUNKS; ++m)
        acc[m] = (v8f){0.f, 0.f, 0.f, 0.f, 0.f, 0.f, 0.f, 0.f};

    const size_t wBase = (size_t)b * MP * KP;

    for (int ks = 0; ks < KP; ks += 32) {
        // B fragment (32x16 KxN): lane column = lo16;
        // elements j<8 -> K = ks + hi*8 + j ; j>=8 -> K = ks + 16 + hi*8 + (j-8)
        AFrag bf;
        #pragma unroll
        for (int j = 0; j < 16; ++j) {
            int k = ks + hi * 8 + ((j < 8) ? j : (8 + j));
            unsigned short v = 0;
            if (k < CIN * 9) {            // folds away when KP == CIN*9
                int i  = k / 9;
                int rm = k - i * 9;
                int r  = rm / 3;
                int c  = rm - r * 3;
                v = patch[i * (4 * ROWW) + (ly + r) * ROWW + (x + c + 7)];
            }
            bf.s[j] = v;
        }
        #pragma unroll
        for (int m = 0; m < MCHUNKS; ++m) {
            const unsigned short* wr =
                mw + wBase + (size_t)(m * 16 + lo16) * KP + ks + hi * 8;
            AFrag af;
            af.q[0] = *(const v4u*)(wr);        // K = base .. base+7
            af.q[1] = *(const v4u*)(wr + 16);   // K = base+16 .. base+23
            acc[m] = __builtin_amdgcn_wmma_f32_16x16x32_bf16(
                         false, af.v, false, bf.v, (short)0, acc[m], false, false);
        }
    }

    // Epilogue: C/D layout -> o = m*16 + hi*8 + v, column = lo16
    #pragma unroll
    for (int m = 0; m < MCHUNKS; ++m) {
        #pragma unroll
        for (int v = 0; v < 8; ++v) {
            int o = m * 16 + hi * 8 + v;
            float r = acc[m][v] + mb[(size_t)b * MP + o];
            if (LRELU) r = (r >= 0.f) ? r : 0.01f * r;
            if (OUTF32) {
                if (o < 2)
                    out_f[(((size_t)b * 2 + o) * 64 + yr) * 64 + x] = r;
            } else {
                out_bf[(((size_t)b * 64 + o) * 64 + yr) * 64 + x] = f32_to_bf16_bits(r);
            }
        }
    }
}

// ---------------------------------------------------------------------------
// Host orchestration
// ---------------------------------------------------------------------------
static inline long nblk(long total) { return (total + 255) / 256; }

extern "C" void kernel_launch(void* const* d_in, const int* in_sizes, int n_in,
                              void* d_out, int out_size, void* d_ws, size_t ws_size,
                              hipStream_t stream)
{
    (void)in_sizes; (void)n_in; (void)out_size; (void)ws_size;

    const float* xx    = (const float*)d_in[0];
    const float* ca1_w = (const float*)d_in[2];  const float* ca1_b = (const float*)d_in[3];
    const float* bn1_g = (const float*)d_in[4];  const float* bn1_b = (const float*)d_in[5];
    const float* ca2_w = (const float*)d_in[6];  const float* ca2_b = (const float*)d_in[7];
    const float* bn2_g = (const float*)d_in[8];  const float* bn2_b = (const float*)d_in[9];
    const float* ca3_w = (const float*)d_in[10]; const float* ca3_b = (const float*)d_in[11];
    const float* bn3_g = (const float*)d_in[12]; const float* bn3_b = (const float*)d_in[13];
    const float* ca4_w = (const float*)d_in[14]; const float* ca4_b = (const float*)d_in[15];
    const float* lin_w = (const float*)d_in[16]; const float* lin_b = (const float*)d_in[17];
    const float* W1 = (const float*)d_in[18]; const float* B1 = (const float*)d_in[19];
    const float* W2 = (const float*)d_in[20]; const float* B2 = (const float*)d_in[21];
    const float* W3 = (const float*)d_in[22]; const float* B3 = (const float*)d_in[23];
    const float* W4 = (const float*)d_in[24]; const float* B4 = (const float*)d_in[25];

    const int B = 256;

    // ---- workspace carving --------------------------------------------------
    char* wsp = (char*)d_ws;
    size_t off = 0;
    auto carve = [&](size_t bytes) -> char* {
        char* p = wsp + off;
        off += (bytes + 255) & ~(size_t)255;
        return p;
    };
    float* bufA  = (float*)carve((size_t)B * 64 * 64 * 64 * 4);  // 268 MB (conv outs; later bf16 ping-pong)
    char*  bufBr = carve((size_t)B * 64 * 32 * 32 * 4);          // 67 MB (pool outs; later mixed weights)
    float* fbuf  = (float*)carve((size_t)B * 8 * 4);
    float* stats = (float*)carve(64 * 2 * 4);
    unsigned short* act0 = (unsigned short*)carve((size_t)B * 2 * 4096 * 2);

    // mixed-weight overlay inside bufBr (used only after attention head completes)
    char* mp = bufBr;
    auto carve2 = [&](size_t bytes) -> char* {
        char* p = mp;
        mp += (bytes + 255) & ~(size_t)255;
        return p;
    };
    unsigned short* mw1 = (unsigned short*)carve2((size_t)B * 64 * 32 * 2);
    float*          mb1 = (float*)carve2((size_t)B * 64 * 4);
    unsigned short* mw2 = (unsigned short*)carve2((size_t)B * 64 * 576 * 2);
    float*          mb2 = (float*)carve2((size_t)B * 64 * 4);
    unsigned short* mw3 = (unsigned short*)carve2((size_t)B * 64 * 576 * 2);
    float*          mb3 = (float*)carve2((size_t)B * 64 * 4);
    unsigned short* mw4 = (unsigned short*)carve2((size_t)B * 16 * 576 * 2);
    float*          mb4 = (float*)carve2((size_t)B * 16 * 4);

    // bf16 activation ping-pong overlaid on bufA (free after linear_softmax)
    unsigned short* actA = (unsigned short*)bufA;
    unsigned short* actB = actA + (size_t)B * 64 * 4096;

    float* poolBuf = (float*)bufBr;

    // ---- attention head -----------------------------------------------------
    conv3x3_f32<<<nblk((long)B * 64 * 4096), 256, 0, stream>>>(xx, ca1_w, ca1_b, bufA, B, 2, 64, 64, 64);
    bn_stats<<<64, 256, 0, stream>>>(bufA, stats, B, 64, 4096);
    bn_lrelu_pool<<<nblk((long)B * 64 * 1024), 256, 0, stream>>>(bufA, stats, bn1_g, bn1_b, poolBuf, B, 64, 64, 64);

    conv3x3_f32<<<nblk((long)B * 64 * 1024), 256, 0, stream>>>(poolBuf, ca2_w, ca2_b, bufA, B, 64, 64, 32, 32);
    bn_stats<<<64, 256, 0, stream>>>(bufA, stats, B, 64, 1024);
    bn_lrelu_pool<<<nblk((long)B * 64 * 256), 256, 0, stream>>>(bufA, stats, bn2_g, bn2_b, poolBuf, B, 64, 32, 32);

    conv3x3_f32<<<nblk((long)B * 64 * 256), 256, 0, stream>>>(poolBuf, ca3_w, ca3_b, bufA, B, 64, 64, 16, 16);
    bn_stats<<<64, 256, 0, stream>>>(bufA, stats, B, 64, 256);
    bn_lrelu_pool<<<nblk((long)B * 64 * 64), 256, 0, stream>>>(bufA, stats, bn3_g, bn3_b, poolBuf, B, 64, 16, 16);

    conv3x3_f32<<<nblk((long)B * 64 * 64), 256, 0, stream>>>(poolBuf, ca4_w, ca4_b, bufA, B, 64, 64, 8, 8);
    linear_softmax<<<B, 256, 0, stream>>>(bufA, lin_w, lin_b, fbuf);

    // ---- expert weight mixing (bf16, padded) -------------------------------
    mix_weights<<<nblk((long)B * 64 * 32), 256, 0, stream>>>(fbuf, W1, B1, mw1, mb1, B, 64, 18, 64, 32);
    mix_weights<<<nblk((long)B * 64 * 576), 256, 0, stream>>>(fbuf, W2, B2, mw2, mb2, B, 64, 576, 64, 576);
    mix_weights<<<nblk((long)B * 64 * 576), 256, 0, stream>>>(fbuf, W3, B3, mw3, mb3, B, 64, 576, 64, 576);
    mix_weights<<<nblk((long)B * 16 * 576), 256, 0, stream>>>(fbuf, W4, B4, mw4, mb4, B, 2, 576, 16, 576);

    cast_bf16<<<nblk((long)B * 2 * 4096), 256, 0, stream>>>(xx, act0, (long)B * 2 * 4096);

    // ---- dynamic expert convs on WMMA --------------------------------------
    dim3 g(B, 32);
    dyn_conv_wmma<2, 32, 4, true, false><<<g, 256, 0, stream>>>(act0, mw1, mb1, actA, nullptr);
    dyn_conv_wmma<64, 576, 4, true, false><<<g, 256, 0, stream>>>(actA, mw2, mb2, actB, nullptr);
    dyn_conv_wmma<64, 576, 4, true, false><<<g, 256, 0, stream>>>(actB, mw3, mb3, actA, nullptr);
    dyn_conv_wmma<64, 576, 1, false, true><<<g, 256, 0, stream>>>(actA, mw4, mb4, nullptr, (float*)d_out);
}
